// NodeMamba3_55860344652207
// MI455X (gfx1250) — matrix operand order repaired and checked by reading, not compiled, not run on hardware
//
#include <hip/hip_runtime.h>
#include <hip/hip_bf16.h>

// ---- problem constants (match reference) ----
#define NN   8192      // nodes / sequence length
#define EE   131072    // edges
#define DIN_ 128
#define DM   128       // d_model
#define DI_  256       // d_inner
#define SS   32        // d_state
#define RR   4         // mimo rank
#define NL_  4

typedef __attribute__((ext_vector_type(16))) _Float16 v16h;
typedef __attribute__((ext_vector_type(8)))  _Float16 v8h;
typedef __attribute__((ext_vector_type(8)))  float    v8f;
typedef __attribute__((ext_vector_type(4)))  int      v4i;

#if __has_builtin(__builtin_amdgcn_global_load_async_to_lds_b128) && \
    __has_builtin(__builtin_amdgcn_s_wait_asynccnt)
#define USE_ASYNC_LDS 1
#else
#define USE_ASYNC_LDS 0
#endif

// ---------------- small helpers ----------------
template <int ACT>
__device__ __forceinline__ float apply_act(float v) {
    if (ACT == 1) return v / (1.0f + __expf(-v));                 // silu
    if (ACT == 2) return (v > 20.0f) ? v : log1pf(__expf(v));     // softplus
    if (ACT == 3) return tanhf(v);                                // tanh
    return v;
}

__global__ void zero_f(float* p, int n) {
    int i = blockIdx.x * blockDim.x + threadIdx.x;
    if (i < n) p[i] = 0.0f;
}

// ---------------- degree + permutation ----------------
__global__ void degree_kernel(const int* __restrict__ ei, float* __restrict__ deg) {
    int e = blockIdx.x * blockDim.x + threadIdx.x;
    if (e < EE) {
        atomicAdd(&deg[ei[e]], 1.0f);       // src
        atomicAdd(&deg[ei[EE + e]], 1.0f);  // dst
    }
}

// stable rank for argsort(-deg): rank_i = #{j : deg[j]>deg[i] || (deg[j]==deg[i] && j<i)}
__global__ void rank_kernel(const float* __restrict__ deg, int* __restrict__ perm) {
    __shared__ float sd[256];
    int i  = blockIdx.x * 256 + threadIdx.x;
    float di = deg[i];
    int cnt = 0;
    for (int j0 = 0; j0 < NN; j0 += 256) {
        sd[threadIdx.x] = deg[j0 + threadIdx.x];
        __syncthreads();
        #pragma unroll 8
        for (int jj = 0; jj < 256; ++jj) {
            int j = j0 + jj;
            float dj = sd[jj];
            cnt += (dj > di) || (dj == di && j < i);
        }
        __syncthreads();
    }
    perm[cnt] = i;  // strict total order -> perm is a permutation
}

// ---------------- layernorm (one row per block, 128 threads) ----------------
__global__ void layernorm_kernel(const float* __restrict__ h,
                                 const float* __restrict__ g,
                                 const float* __restrict__ b,
                                 float* __restrict__ hn) {
    __shared__ float s[128];
    int row = blockIdx.x, d = threadIdx.x;
    float x = h[(size_t)row * DM + d];
    s[d] = x; __syncthreads();
    for (int o = 64; o > 0; o >>= 1) { if (d < o) s[d] += s[d + o]; __syncthreads(); }
    float mu = s[0] * (1.0f / DM); __syncthreads();
    float c = x - mu;
    s[d] = c * c; __syncthreads();
    for (int o = 64; o > 0; o >>= 1) { if (d < o) s[d] += s[d + o]; __syncthreads(); }
    float var = s[0] * (1.0f / DM);
    hn[(size_t)row * DM + d] = c * __frsqrt_rn(var + 1e-5f) * g[d] + b[d];
}

// ---------------- templated WMMA GEMM ----------------
// out = act(rowmap(A) @ W + bias [+ resid]).
// block = 128 threads (4 waves); each wave: 16(M) x 64(N); block tile 64x64.
// grid = (Nc/64, NN/64). K % 32 == 0 (compile-time).
template <int K, int ACT, bool HASBIAS, bool HASRESID, bool HASMAP>
__global__ __launch_bounds__(128)
void gemm_wmma(const float* __restrict__ A, int lda, const int* __restrict__ rowmap,
               const float* __restrict__ W, int ldw,
               const float* __restrict__ bias,
               const float* __restrict__ resid, int ldr,
               float* __restrict__ out, int ldo) {
    __shared__ __align__(64) _Float16 sW[64][32];   // [n][k] : 4 KB per k-step

    const int tid  = threadIdx.x;
    const int wave = tid >> 5;
    const int lane = tid & 31;
    const int half = lane >> 4;   // 0: lanes 0-15, 1: lanes 16-31
    const int m16  = lane & 15;
    const int rowM = blockIdx.y * 64 + wave * 16;
    const int col0 = blockIdx.x * 64;

    const int arow = rowM + m16;
    const int grow = HASMAP ? rowmap[arow] : arow;
    const float* Arow = A + (size_t)grow * lda;

    // cooperative W staging: thread covers k = tid>>2, n in [sn0, sn0+16)
    const int sk  = tid >> 2;
    const int sn0 = (tid & 3) * 16;

    v8f acc[4] = {};

    #pragma unroll 2
    for (int k0 = 0; k0 < K; k0 += 32) {
        // -------- stage W[k0+sk][col0+sn0 .. +15] -> sW (f32 -> f16) --------
        const float* wrow = W + (size_t)(k0 + sk) * ldw + col0 + sn0;
        union { float4 q[4]; float f[16]; } uw;
        uw.q[0] = ((const float4*)wrow)[0];
        uw.q[1] = ((const float4*)wrow)[1];
        uw.q[2] = ((const float4*)wrow)[2];
        uw.q[3] = ((const float4*)wrow)[3];
        __syncthreads();   // previous iteration's consumers done
        #pragma unroll
        for (int e = 0; e < 16; ++e) sW[sn0 + e][sk] = (_Float16)uw.f[e];

        // -------- A fragment (ISA 16-bit A layout): K = (e/8)*16 + half*8 + e%8
        const float4* Ap = (const float4*)(Arow + k0 + half * 8);
        float4 q0 = Ap[0], q1 = Ap[1], q2 = Ap[4], q3 = Ap[5];
        v16h a;
        a[0]=(_Float16)q0.x;  a[1]=(_Float16)q0.y;  a[2]=(_Float16)q0.z;  a[3]=(_Float16)q0.w;
        a[4]=(_Float16)q1.x;  a[5]=(_Float16)q1.y;  a[6]=(_Float16)q1.z;  a[7]=(_Float16)q1.w;
        a[8]=(_Float16)q2.x;  a[9]=(_Float16)q2.y;  a[10]=(_Float16)q2.z; a[11]=(_Float16)q2.w;
        a[12]=(_Float16)q3.x; a[13]=(_Float16)q3.y; a[14]=(_Float16)q3.z; a[15]=(_Float16)q3.w;

        __syncthreads();   // sW visible

        // -------- 4 x WMMA, A fragment reused --------
        #pragma unroll
        for (int j = 0; j < 4; ++j) {
            // B layout: lane holds column n = j*16+m16; K = (e/8)*16 + half*8 + e%8
            const _Float16* bp = &sW[j * 16 + m16][half * 8];
            v8h blo = *(const v8h*)bp;
            v8h bhi = *(const v8h*)(bp + 16);
            v16h bfr = __builtin_shufflevector(blo, bhi,
                0,1,2,3,4,5,6,7,8,9,10,11,12,13,14,15);
            acc[j] = __builtin_amdgcn_wmma_f32_16x16x32_f16(false, a, false, bfr,
                                                            (short)0, acc[j], false, false);
        }
    }

    // -------- epilogue --------
    #pragma unroll
    for (int j = 0; j < 4; ++j) {
        const int n = col0 + j * 16 + m16;
        const float bv = HASBIAS ? bias[n] : 0.0f;
        #pragma unroll
        for (int r = 0; r < 8; ++r) {
            int row = rowM + half * 8 + r;   // C/D layout: M = r + 8*half
            float v = acc[j][r] + bv;
            if (HASRESID) v += resid[(size_t)row * ldr + n];
            out[(size_t)row * ldo + n] = apply_act<ACT>(v);
        }
    }
}

// ---------------- bidirectional selective scan ----------------
// grid = 2 (x: 0=forward, 1=backward), block = 1024. thread = (r = tid>>8, d = tid&255).
// State h[r,d,s] (s=0..31) lives in registers for the whole 8192-step scan.
__global__ __launch_bounds__(1024)
void scan_kernel(const float* __restrict__ u,  const float* __restrict__ dt,
                 const float* __restrict__ Bm, const float* __restrict__ Cm,
                 const float* __restrict__ A_log,
                 float* __restrict__ yf, float* __restrict__ yb) {
    __shared__ __align__(16) float sB[RR * SS];   // 128
    __shared__ __align__(16) float sC[RR * SS];   // 128
    __shared__ float sP[1024];

    const int tid = threadIdx.x;
    const int d = tid & 255;
    const int r = tid >> 8;
    const bool rev = (blockIdx.x != 0);
    float* y = rev ? yb : yf;

    float Ar[SS], hst[SS];
    #pragma unroll
    for (int s = 0; s < SS; ++s) {
        Ar[s]  = -__expf(A_log[d * SS + s]);
        hst[s] = 0.0f;
    }

    const int rb = r * SS;
    for (int t = 0; t < NN; ++t) {
        const int tt = rev ? (NN - 1 - t) : t;

#if USE_ASYNC_LDS
        // gfx1250 async copy: ASYNCcnt-tracked global->LDS, 16B per lane
        if (tid < 64) {
            int q = tid & 31;
            const float* src = (tid < 32) ? (Bm + (size_t)tt * (RR * SS) + q * 4)
                                          : (Cm + (size_t)tt * (RR * SS) + q * 4);
            float* dst = (tid < 32) ? (sB + q * 4) : (sC + q * 4);
            __builtin_amdgcn_global_load_async_to_lds_b128(
                (__attribute__((address_space(1))) v4i*)(uintptr_t)src,
                (__attribute__((address_space(3))) v4i*)(uintptr_t)dst,
                0, 0);
            __builtin_amdgcn_s_wait_asynccnt(0);
        }
#else
        if (tid < 128)        sB[tid]       = Bm[(size_t)tt * (RR * SS) + tid];
        else if (tid < 256)   sC[tid - 128] = Cm[(size_t)tt * (RR * SS) + (tid - 128)];
#endif
        __syncthreads();

        const float dtv = dt[(size_t)tt * DI_ + d];
        const float uv  = u [(size_t)tt * DI_ + d];
        const float du  = dtv * uv;

        // prefetch next step's operands (global_prefetch_b8)
        if (t + 1 < NN) {
            const int tn = rev ? (tt - 1) : (tt + 1);
            __builtin_prefetch(&dt[(size_t)tn * DI_ + d], 0, 1);
            __builtin_prefetch(&u [(size_t)tn * DI_ + d], 0, 1);
            if (tid < 128) __builtin_prefetch(&Bm[(size_t)tn * (RR * SS) + tid], 0, 1);
            else if (tid < 256) __builtin_prefetch(&Cm[(size_t)tn * (RR * SS) + (tid - 128)], 0, 1);
        }

        float p = 0.0f;
        #pragma unroll
        for (int s = 0; s < SS; ++s) {
            float dA = __expf(dtv * Ar[s]);
            hst[s] = hst[s] * dA + sB[rb + s] * du;
            p += hst[s] * sC[rb + s];
        }
        sP[tid] = p;
        __syncthreads();
        if (r == 0)
            y[(size_t)tt * DI_ + d] = sP[d] + sP[256 + d] + sP[512 + d] + sP[768 + d];
        __syncthreads();  // protect sB/sC/sP before next step's writes
    }
}

// ---------------- gated combine: yc = (yf + yb + Dskip*u) * silu(z) ----------------
__global__ void ycomb_kernel(const float* __restrict__ yf, const float* __restrict__ yb,
                             const float* __restrict__ u,  const float* __restrict__ z,
                             const float* __restrict__ Dskip, float* __restrict__ yc) {
    int i = blockIdx.x * blockDim.x + threadIdx.x;
    if (i < NN * DI_) {
        int d = i & (DI_ - 1);
        float zv = z[i];
        float sz = zv / (1.0f + __expf(-zv));
        yc[i] = (yf[i] + yb[i] + Dskip[d] * u[i]) * sz;
    }
}

// ---------------- attention pool ----------------
__global__ void scores_kernel(const float* __restrict__ tpool, const float* __restrict__ pool_v,
                              float* __restrict__ scores) {
    int i = blockIdx.x * blockDim.x + threadIdx.x;
    if (i < NN) {
        float acc = 0.0f;
        const float* row = tpool + (size_t)i * DM;
        #pragma unroll 8
        for (int d = 0; d < DM; ++d) acc += row[d] * pool_v[d];
        scores[i] = acc;
    }
}

__global__ __launch_bounds__(1024)
void softmax_stats_kernel(const float* __restrict__ scores, float* __restrict__ red) {
    __shared__ float s[1024];
    int tid = threadIdx.x;
    float mx = -3.402823e38f;
    for (int i = tid; i < NN; i += 1024) mx = fmaxf(mx, scores[i]);
    s[tid] = mx; __syncthreads();
    for (int o = 512; o > 0; o >>= 1) { if (tid < o) s[tid] = fmaxf(s[tid], s[tid + o]); __syncthreads(); }
    mx = s[0]; __syncthreads();
    float sm = 0.0f;
    for (int i = tid; i < NN; i += 1024) sm += __expf(scores[i] - mx);
    s[tid] = sm; __syncthreads();
    for (int o = 512; o > 0; o >>= 1) { if (tid < o) s[tid] += s[tid + o]; __syncthreads(); }
    if (tid == 0) { red[0] = mx; red[1] = s[0]; }
}

__global__ void pool_weight_kernel(float* __restrict__ scores, const float* __restrict__ red) {
    int i = blockIdx.x * blockDim.x + threadIdx.x;
    if (i < NN) scores[i] = __expf(scores[i] - red[0]) / red[1];
}

// grid=64, block=128: part[b*128+d] = sum over rows of block b of w_i * h[i,d]
__global__ void pool_partial_kernel(const float* __restrict__ w, const float* __restrict__ h,
                                    float* __restrict__ part) {
    int b = blockIdx.x, d = threadIdx.x;
    float acc = 0.0f;
    int i0 = b * 128;
    for (int i = i0; i < i0 + 128; ++i) acc += w[i] * h[(size_t)i * DM + d];
    part[b * 128 + d] = acc;
}

__global__ void final_kernel(const float* __restrict__ part, const float* __restrict__ head_W,
                             const float* __restrict__ head_b, float* __restrict__ out) {
    __shared__ float s[128];
    int d = threadIdx.x;
    float acc = 0.0f;
    for (int b = 0; b < 64; ++b) acc += part[b * 128 + d];
    s[d] = acc * head_W[d]; __syncthreads();
    for (int o = 64; o > 0; o >>= 1) { if (d < o) s[d] += s[d + o]; __syncthreads(); }
    if (d == 0) out[0] = s[0] + head_b[0];
}

// ---------------- host orchestration ----------------
extern "C" void kernel_launch(void* const* d_in, const int* in_sizes, int n_in,
                              void* d_out, int out_size, void* d_ws, size_t ws_size,
                              hipStream_t stream) {
    const float* x       = (const float*)d_in[0];
    const int*   ei      = (const int*)  d_in[1];
    const float* W_embed = (const float*)d_in[2];
    const float* b_embed = (const float*)d_in[3];
    const float* ln_g    = (const float*)d_in[4];
    const float* ln_b    = (const float*)d_in[5];
    const float* W_in    = (const float*)d_in[6];
    const float* W_dt    = (const float*)d_in[7];
    const float* b_dt    = (const float*)d_in[8];
    const float* W_B     = (const float*)d_in[9];
    const float* W_C     = (const float*)d_in[10];
    const float* A_log   = (const float*)d_in[11];
    const float* Dskip   = (const float*)d_in[12];
    const float* W_out   = (const float*)d_in[13];
    const float* pool_W  = (const float*)d_in[14];
    const float* pool_v  = (const float*)d_in[15];
    const float* head_W  = (const float*)d_in[16];
    const float* head_b  = (const float*)d_in[17];

    // workspace carve-up
    float* deg  = (float*)d_ws;                 // NN
    int*   perm = (int*)(deg + NN);             // NN
    float* h    = (float*)(perm + NN);          // NN*DM
    float* hn   = h  + (size_t)NN * DM;         // NN*DM   (aliased as yc later)
    float* u    = hn + (size_t)NN * DM;         // NN*DI   (aliased as tpool later)
    float* z    = u  + (size_t)NN * DI_;        // NN*DI
    float* dt   = z  + (size_t)NN * DI_;        // NN*DI
    float* Bm   = dt + (size_t)NN * DI_;        // NN*128
    float* Cm   = Bm + (size_t)NN * RR * SS;    // NN*128
    float* yfb  = Cm + (size_t)NN * RR * SS;    // NN*DI (fwd)
    float* ybb  = yfb + (size_t)NN * DI_;       // NN*DI (bwd)
    float* red  = ybb + (size_t)NN * DI_;       // 2
    float* part = red + 2;                      // 64*128
    float* scores = part + 64 * 128;            // NN
    float* yc    = hn;                          // alias (hn dead after B/C GEMMs)
    float* tpool = u;                           // alias (u dead at pool time)

    // 1. degree + stable descending argsort
    zero_f<<<(NN + 255) / 256, 256, 0, stream>>>(deg, NN);
    degree_kernel<<<(EE + 255) / 256, 256, 0, stream>>>(ei, deg);
    rank_kernel<<<NN / 256, 256, 0, stream>>>(deg, perm);

    // 2. embed: h = x[perm] @ W_embed + b_embed   (K=128, Nc=128)
    gemm_wmma<128, 0, true, false, true><<<dim3(DM / 64, NN / 64), 128, 0, stream>>>(
        x, DIN_, perm, W_embed, DM, b_embed, nullptr, 0, h, DM);

    // 3. layers
    for (int i = 0; i < NL_; ++i) {
        layernorm_kernel<<<NN, 128, 0, stream>>>(h, ln_g + i * DM, ln_b + i * DM, hn);

        const float* Wi = W_in + (size_t)i * DM * (2 * DI_);
        // u = silu(hn @ W_in[:, :DI])
        gemm_wmma<128, 1, false, false, false><<<dim3(DI_ / 64, NN / 64), 128, 0, stream>>>(
            hn, DM, nullptr, Wi, 2 * DI_, nullptr, nullptr, 0, u, DI_);
        // z = hn @ W_in[:, DI:]
        gemm_wmma<128, 0, false, false, false><<<dim3(DI_ / 64, NN / 64), 128, 0, stream>>>(
            hn, DM, nullptr, Wi + DI_, 2 * DI_, nullptr, nullptr, 0, z, DI_);
        // dt = softplus(u @ W_dt + b_dt)
        gemm_wmma<256, 2, true, false, false><<<dim3(DI_ / 64, NN / 64), 128, 0, stream>>>(
            u, DI_, nullptr, W_dt + (size_t)i * DI_ * DI_, DI_, b_dt + i * DI_,
            nullptr, 0, dt, DI_);
        // B = u @ W_B ; C = u @ W_C   (Nc = R*S = 128)
        gemm_wmma<256, 0, false, false, false><<<dim3((RR * SS) / 64, NN / 64), 128, 0, stream>>>(
            u, DI_, nullptr, W_B + (size_t)i * DI_ * RR * SS, RR * SS,
            nullptr, nullptr, 0, Bm, RR * SS);
        gemm_wmma<256, 0, false, false, false><<<dim3((RR * SS) / 64, NN / 64), 128, 0, stream>>>(
            u, DI_, nullptr, W_C + (size_t)i * DI_ * RR * SS, RR * SS,
            nullptr, nullptr, 0, Cm, RR * SS);

        // bidirectional selective scan (blockIdx.x: 0=fwd, 1=bwd)
        scan_kernel<<<2, 1024, 0, stream>>>(u, dt, Bm, Cm,
                                            A_log + (size_t)i * DI_ * SS, yfb, ybb);

        // yc = (yf + yb + Dskip*u) * silu(z)
        ycomb_kernel<<<(NN * DI_ + 255) / 256, 256, 0, stream>>>(
            yfb, ybb, u, z, Dskip + i * DI_, yc);

        // h = h + yc @ W_out
        gemm_wmma<256, 0, false, true, false><<<dim3(DM / 64, NN / 64), 128, 0, stream>>>(
            yc, DI_, nullptr, W_out + (size_t)i * DI_ * DM, DM,
            nullptr, h, DM, h, DM);
    }

    // 4. attention pool + head
    gemm_wmma<128, 3, false, false, false><<<dim3(DM / 64, NN / 64), 128, 0, stream>>>(
        h, DM, nullptr, pool_W, DM, nullptr, nullptr, 0, tpool, DM);
    scores_kernel<<<NN / 256, 256, 0, stream>>>(tpool, pool_v, scores);
    softmax_stats_kernel<<<1, 1024, 0, stream>>>(scores, red);
    pool_weight_kernel<<<NN / 256, 256, 0, stream>>>(scores, red);
    pool_partial_kernel<<<64, 128, 0, stream>>>(scores, h, part);
    final_kernel<<<1, 128, 0, stream>>>(part, head_W, head_b, (float*)d_out);
}
